// LearnedClassVectors_65197603554143
// MI455X (gfx1250) — matrix operand order repaired and checked by reading, not compiled, not run on hardware
//
#include <hip/hip_runtime.h>

// ---------------------------------------------------------------------------
// LearnedClassVectors on MI455X (gfx1250, wave32, WMMA).
//
// out[b,o,gd,gh,gw] = sum_f W[o,f] * vectors[bucket(x_voxel(f))][v(f)] + bias[o]
//   GEMM: M-dim = O = 768 (rows of fc_w), N-dim = 65536 patches, K = 512.
//
// Fused: bucketize+embed feature tiles straight into LDS, f16 WMMA 16x16x32
// with f32 accumulation, async global->LDS staging for the W tiles,
// coalesced stores along the patch dim.
// ---------------------------------------------------------------------------

typedef _Float16 half_t;
typedef __attribute__((ext_vector_type(8)))  half_t v8h;
typedef __attribute__((ext_vector_type(16))) half_t v16h;
typedef __attribute__((ext_vector_type(8)))  float  v8f;
typedef int v4i __attribute__((vector_size(16)));   // matches builtin param type

#define AS1 __attribute__((address_space(1)))
#define AS3 __attribute__((address_space(3)))

#if __has_builtin(__builtin_amdgcn_global_load_async_to_lds_b128) && \
    __has_builtin(__builtin_amdgcn_s_wait_asynccnt)
#define HAVE_ASYNC_LDS 1
#else
#define HAVE_ASYNC_LDS 0
#endif

#define OUT_DIM   768
#define KDIM      512
#define PPB       32768      // patches per batch (32^3)
#define M_TILES   512        // 65536 / 128
#define O_TILES   6          // 768 / 128
#define LDST      40         // LDS row stride in halfs (80B, bank-friendly)
#define NVEC      104        // 13 * 8 embedding table elements

__device__ __forceinline__ int hu_bucket(float x) {
    // searchsorted(HU_INTERVALS, x, side='right') == #{k : iv[k] <= x}
    const float hu[12] = {-1000.f, -900.f, -400.f, -100.f, -50.f, -10.f,
                          20.f, 40.f, 60.f, 100.f, 800.f, 1000.f};
    int idx = 0;
#pragma unroll
    for (int k = 0; k < 12; ++k) idx += (x >= hu[k]) ? 1 : 0;
    return idx;
}

__global__ void convert_f16(const float* __restrict__ w,
                            const float* __restrict__ vec,
                            half_t* __restrict__ wh,
                            half_t* __restrict__ vtab, int nW) {
    int i = blockIdx.x * blockDim.x + threadIdx.x;
    if (i < nW)              wh[i]        = (half_t)w[i];
    else if (i < nW + NVEC)  vtab[i - nW] = (half_t)vec[i - nW];
}

__global__ __launch_bounds__(256) void lcv_gemm(
    const float* __restrict__ x,        // [2,1,128,128,128]
    const half_t* __restrict__ vtab,    // [13,8] f16 (L2-resident)
    const half_t* __restrict__ wh,      // [768,512] f16
    const float* __restrict__ bias,     // [768]
    float* __restrict__ out)            // [2,768,32,32,32]
{
    __shared__ __align__(16) half_t Ws[2][128][LDST];   // W tile:   128 o-rows x 32 k
    __shared__ __align__(16) half_t Fs[2][128][LDST];   // feat tile: 128 m-rows x 32 k
    __shared__ float bias_s[128];

    const int tid   = threadIdx.x;
    const int mBase = blockIdx.x * 128;   // over all 65536 patches
    const int oBase = blockIdx.y * 128;   // over 768 outputs

    // decompose patch base: m = b*32768 + gd*1024 + gh*32 + gw
    const int b       = mBase >> 15;
    const int mWithin = mBase & 32767;
    const int gd      = mWithin >> 10;
    const int gh0     = (mWithin >> 5) & 31;   // multiple of 4

    if (tid < 128) bias_s[tid] = bias[oBase + tid];

    // --- staging helpers ----------------------------------------------------
    auto stageW = [&](int buf, int kt) {
#pragma unroll
        for (int i = 0; i < 2; ++i) {
            int u   = tid + i * 256;       // 512 units of 16B
            int r   = u >> 2;
            int seg = u & 3;
            const half_t* src = wh + (size_t)(oBase + r) * KDIM + kt * 32 + seg * 8;
            half_t*       dst = &Ws[buf][r][seg * 8];
#if HAVE_ASYNC_LDS
            // ASYNCcnt-tracked DMA: no VGPR round trip, overlaps with WMMA
            __builtin_amdgcn_global_load_async_to_lds_b128(
                (AS1 v4i*)(AS1 void*)(void*)src,
                (AS3 v4i*)(AS3 void*)(void*)dst, 0, 0);
#else
            *(v8h*)dst = *(const v8h*)src;
#endif
        }
    };
    auto stageF = [&](int buf, int kt) {
        // k-tile kt covers feature cols ((pd*4+ph)*4+pw)*8+v with pd*4+ph == kt
        const int pd  = kt >> 2, ph = kt & 3;
        const int r   = tid >> 1;          // patch row within tile
        const int pw0 = (tid & 1) * 2;     // this thread: voxels pw0, pw0+1
        const int gh  = gh0 + (r >> 5);
        const int gw  = r & 31;
        const int d   = gd * 4 + pd;
        const int h   = gh * 4 + ph;
        const int w   = gw * 4 + pw0;
        const float2 xv = *(const float2*)(x + (((size_t)b * 128 + d) * 128 + h) * 128 + w);
        const int i0 = hu_bucket(xv.x);
        const int i1 = hu_bucket(xv.y);
        // 16B cached gathers from the 208B f16 table (LOADcnt, fully overlapped)
        v8h e0 = *(const v8h*)(vtab + i0 * 8);
        v8h e1 = *(const v8h*)(vtab + i1 * 8);
        *(v8h*)&Fs[buf][r][pw0 * 8 + 0] = e0;
        *(v8h*)&Fs[buf][r][pw0 * 8 + 8] = e1;
    };

    // --- wave tiling: 8 waves, each 32(O) x 64(m) ---------------------------
    const int lane  = tid & 31;
    const int wv    = tid >> 5;
    const int waveO = wv & 3;        // *32 rows of O
    const int waveM = wv >> 2;       // *64 cols of m
    const int lrow  = lane & 15;
    const int lhalf = lane >> 4;

    v8f acc[2][4];
#pragma unroll
    for (int i = 0; i < 2; ++i)
#pragma unroll
        for (int j = 0; j < 4; ++j) acc[i][j] = (v8f){0,0,0,0,0,0,0,0};

    stageW(0, 0);
    stageF(0, 0);
#if HAVE_ASYNC_LDS
    __builtin_amdgcn_s_wait_asynccnt(0);
#endif
    __syncthreads();

    int buf = 0;
    for (int kt = 0; kt < 16; ++kt) {
        if (kt < 15) {                 // prefetch next k-slab into other buffer
            stageW(buf ^ 1, kt + 1);
            stageF(buf ^ 1, kt + 1);
        }

        // A fragments: 16x32 f16, rows = o. Lane L row = base+(L&15);
        // elems 0..7 = K c0..c0+7, elems 8..15 = K c0+16..c0+23, c0=(L>=16)*8.
        const int c0 = lhalf * 8;
        v16h a[2];
#pragma unroll
        for (int i = 0; i < 2; ++i) {
            int r  = waveO * 32 + i * 16 + lrow;
            v8h lo = *(const v8h*)&Ws[buf][r][c0];
            v8h hi = *(const v8h*)&Ws[buf][r][c0 + 16];
            a[i] = __builtin_shufflevector(lo, hi, 0,1,2,3,4,5,6,7,
                                                   8,9,10,11,12,13,14,15);
        }
        // B fragments: 32x16 f16, cols = m. Lane L col = base+(L&15);
        // elems 0..7 = K kc..kc+7, elems 8..15 = K kc+16..kc+23, kc=(L>=16)*8.
        const int kc = lhalf * 8;
        v16h f[4];
#pragma unroll
        for (int j = 0; j < 4; ++j) {
            int c  = waveM * 64 + j * 16 + lrow;
            v8h lo = *(const v8h*)&Fs[buf][c][kc];
            v8h hi = *(const v8h*)&Fs[buf][c][kc + 16];
            f[j] = __builtin_shufflevector(lo, hi, 0,1,2,3,4,5,6,7,
                                                   8,9,10,11,12,13,14,15);
        }

#pragma unroll
        for (int i = 0; i < 2; ++i)
#pragma unroll
            for (int j = 0; j < 4; ++j)
                acc[i][j] = __builtin_amdgcn_wmma_f32_16x16x32_f16(
                    false, a[i], false, f[j], (short)0, acc[i][j], false, false);

#if HAVE_ASYNC_LDS
        __builtin_amdgcn_s_wait_asynccnt(0);   // drain async stage of buf^1
#endif
        __syncthreads();
        buf ^= 1;
    }

    // --- epilogue: D elem e of lane L -> o_local = e + 8*(L>=16), m_local = L&15
    // out[b][o][m] contiguous in m -> consecutive lanes write consecutive floats.
#pragma unroll
    for (int i = 0; i < 2; ++i) {
#pragma unroll
        for (int j = 0; j < 4; ++j) {
            const int oL0 = waveO * 32 + i * 16 + lhalf * 8;
            const int mL  = waveM * 64 + j * 16 + lrow;
#pragma unroll
            for (int e = 0; e < 8; ++e) {
                const int oL = oL0 + e;
                const size_t off = ((size_t)(b * OUT_DIM + oBase + oL)) * PPB
                                   + (size_t)(mWithin + mL);
                out[off] = acc[i][j][e] + bias_s[oL];
            }
        }
    }
}

// ---------------------------------------------------------------------------
extern "C" void kernel_launch(void* const* d_in, const int* in_sizes, int n_in,
                              void* d_out, int out_size, void* d_ws, size_t ws_size,
                              hipStream_t stream) {
    const float* x       = (const float*)d_in[0];
    const float* vectors = (const float*)d_in[1];
    const float* fc_w    = (const float*)d_in[2];
    const float* fc_b    = (const float*)d_in[3];
    float*  out  = (float*)d_out;
    half_t* wh   = (half_t*)d_ws;                          // 768 KB
    half_t* vtab = (half_t*)d_ws + (size_t)OUT_DIM * KDIM; // 208 B

    const int nW = OUT_DIM * KDIM;
    convert_f16<<<(nW + NVEC + 255) / 256, 256, 0, stream>>>(fc_w, vectors, wh, vtab, nW);

    dim3 grid(M_TILES, O_TILES);                 // 512 x 6 blocks
    lcv_gemm<<<grid, 256, 0, stream>>>(x, vtab, wh, fc_b, out);
}
